// SelfAttention_73297911873724
// MI455X (gfx1250) — matrix operand order repaired and checked by reading, not compiled, not run on hardware
//
#include <hip/hip_runtime.h>
#include <hip/hip_bf16.h>

// ---------------------------------------------------------------------------
// Transformer block for MI455X (gfx1250): all matmuls via v_wmma_f32_16x16x32_f16,
// flash-attention schedule (no materialized NxN scores), f32 accumulation.
// GEMMs use a 32x64 wave tile: 8 WMMAs per 12 b128 loads (B frags shared by
// two A-row fragments) to keep the matrix pipe ahead of vmem issue.
// ---------------------------------------------------------------------------

typedef __attribute__((ext_vector_type(16))) _Float16 v16h;
typedef __attribute__((ext_vector_type(8)))  _Float16 v8h;
typedef __attribute__((ext_vector_type(4)))  _Float16 v4h;
typedef __attribute__((ext_vector_type(8)))  float    v8f;

#define NB    2
#define NSEQ  4096
#define DIM   512
#define NHEAD 8
#define HD    64
#define HID   2048
#define ROWS  (NB * NSEQ)          // 8192 tokens

__device__ __forceinline__ v8f wmma_f16(v16h a, v16h b, v8f c) {
  return __builtin_amdgcn_wmma_f32_16x16x32_f16(false, a, false, b, (short)0, c,
                                                false, false);
}

// A-fragment (16x32 f16): lane l holds row (l%16); element j maps to
// K = (l/16)*16 + j  -> two contiguous 16B runs.
__device__ __forceinline__ v16h load_a_frag(const _Float16* row, int k, int hl) {
  v8h lo = *(const v8h*)(row + k + hl * 16);
  v8h hi = *(const v8h*)(row + k + hl * 16 + 8);
  return __builtin_shufflevector(lo, hi, 0, 1, 2, 3, 4, 5, 6, 7,
                                 8, 9, 10, 11, 12, 13, 14, 15);
}

// ---------------------------------------------------------------------------
// Wave GEMM: 32(M) x 64(N) tile. A [M x KD] row-major f16, Bt [N x KD]
// (B transposed: a B-fragment column is 32 contiguous bytes).
// Per k-step: 4 A-loads + 8 B-loads feed 8 WMMAs.
// ---------------------------------------------------------------------------
template <int KD>
__device__ __forceinline__ void gemm_wave_32x64(const _Float16* __restrict__ Arow0,
                                                const _Float16* __restrict__ Arow1,
                                                const _Float16* __restrict__ Bt,
                                                int n0, int hl, int ln,
                                                v8f acc0[4], v8f acc1[4]) {
  const _Float16* b0 = Bt + (size_t)(n0 + 0 + ln) * KD;
  const _Float16* b1 = Bt + (size_t)(n0 + 16 + ln) * KD;
  const _Float16* b2 = Bt + (size_t)(n0 + 32 + ln) * KD;
  const _Float16* b3 = Bt + (size_t)(n0 + 48 + ln) * KD;
#pragma unroll 4
  for (int k = 0; k < KD; k += 32) {
    v16h a0 = load_a_frag(Arow0, k, hl);
    v16h a1 = load_a_frag(Arow1, k, hl);
    v16h bb0 = *(const v16h*)(b0 + k + hl * 16);
    v16h bb1 = *(const v16h*)(b1 + k + hl * 16);
    v16h bb2 = *(const v16h*)(b2 + k + hl * 16);
    v16h bb3 = *(const v16h*)(b3 + k + hl * 16);
    acc0[0] = wmma_f16(a0, bb0, acc0[0]);
    acc1[0] = wmma_f16(a1, bb0, acc1[0]);
    acc0[1] = wmma_f16(a0, bb1, acc0[1]);
    acc1[1] = wmma_f16(a1, bb1, acc1[1]);
    acc0[2] = wmma_f16(a0, bb2, acc0[2]);
    acc1[2] = wmma_f16(a1, bb2, acc1[2]);
    acc0[3] = wmma_f16(a0, bb3, acc0[3]);
    acc1[3] = wmma_f16(a1, bb3, acc1[3]);
  }
}

// ---------------------------------------------------------------------------
// Weight transpose + f32 -> f16 convert (one-time, tiny).
// ---------------------------------------------------------------------------
__global__ void convert_w_kernel(const float* __restrict__ wqkv,
                                 const float* __restrict__ wproj,
                                 const float* __restrict__ w1,
                                 const float* __restrict__ w2,
                                 _Float16* __restrict__ tqkv,
                                 _Float16* __restrict__ tproj,
                                 _Float16* __restrict__ t1,
                                 _Float16* __restrict__ t2) {
  int idx = blockIdx.x * 256 + threadIdx.x;
  if (idx < 786432) {                               // [1536 x 512]
    int n = idx >> 9, k = idx & 511;
    tqkv[idx] = (_Float16)wqkv[(size_t)k * 1536 + n];
  } else if (idx < 1048576) {                       // [512 x 512]
    int j = idx - 786432;
    int n = j >> 9, k = j & 511;
    tproj[j] = (_Float16)wproj[(size_t)k * 512 + n];
  } else if (idx < 2097152) {                       // [2048 x 512]
    int j = idx - 1048576;
    int n = j >> 9, k = j & 511;
    t1[j] = (_Float16)w1[(size_t)k * 2048 + n];
  } else if (idx < 3145728) {                       // [512 x 2048]
    int j = idx - 2097152;
    int n = j >> 11, k = j & 2047;
    t2[j] = (_Float16)w2[(size_t)k * 512 + n];
  }
}

// ---------------------------------------------------------------------------
// LayerNorm: one wave per 512-element row, f32 in -> f16 out.
// ---------------------------------------------------------------------------
__global__ void ln_kernel(const float* __restrict__ X, const float* __restrict__ g,
                          const float* __restrict__ bta, _Float16* __restrict__ Y) {
  int row  = blockIdx.x * 8 + (threadIdx.x >> 5);
  int lane = threadIdx.x & 31;
  const float* xr = X + (size_t)row * DIM;
  float vals[16];
  float s = 0.f, ss = 0.f;
#pragma unroll
  for (int c = 0; c < 4; c++) {
    float4 f = *(const float4*)(xr + c * 128 + lane * 4);
    vals[c * 4 + 0] = f.x; vals[c * 4 + 1] = f.y;
    vals[c * 4 + 2] = f.z; vals[c * 4 + 3] = f.w;
    s  += f.x + f.y + f.z + f.w;
    ss += f.x * f.x + f.y * f.y + f.z * f.z + f.w * f.w;
  }
#pragma unroll
  for (int m = 1; m < 32; m <<= 1) {
    s  += __shfl_xor(s, m);
    ss += __shfl_xor(ss, m);
  }
  float mu  = s * (1.0f / DIM);
  float var = ss * (1.0f / DIM) - mu * mu;
  float rs  = rsqrtf(var + 1e-5f);
#pragma unroll
  for (int c = 0; c < 4; c++) {
    v4h o;
#pragma unroll
    for (int i = 0; i < 4; i++) {
      int col = c * 128 + lane * 4 + i;
      o[i] = (_Float16)((vals[c * 4 + i] - mu) * rs * g[col] + bta[col]);
    }
    *(v4h*)(Y + (size_t)row * DIM + c * 128 + lane * 4) = o;
  }
}

// ---------------------------------------------------------------------------
// QKV GEMM (32x64 tile): xln @ WqkvT -> Q [bh,n,hd], K [bh,n,hd], VT [bh,hd,n].
// VT rows are written as packed 16B stores (token axis contiguous per lane).
// ---------------------------------------------------------------------------
__global__ void gemm_qkv_kernel(const _Float16* __restrict__ XLN,
                                const _Float16* __restrict__ Wt,
                                const float* __restrict__ bias,
                                _Float16* __restrict__ Qb, _Float16* __restrict__ Kb,
                                _Float16* __restrict__ VTb) {
  int wave = blockIdx.x * 8 + (threadIdx.x >> 5);
  const int NT = (3 * DIM) / 64;   // 24
  int mt = wave / NT, n0 = (wave % NT) * 64;
  int lane = threadIdx.x & 31, hl = lane >> 4, ln = lane & 15;
  int m0 = mt * 32;
  v8f acc[2][4] = {};
  gemm_wave_32x64<DIM>(XLN + (size_t)(m0 + ln) * DIM,
                       XLN + (size_t)(m0 + 16 + ln) * DIM,
                       Wt, n0, hl, ln, acc[0], acc[1]);
#pragma unroll
  for (int s = 0; s < 2; s++) {
#pragma unroll
    for (int nt = 0; nt < 4; nt++) {
      int c = n0 + nt * 16 + ln;
      int seg = c >> 9, cc = c & 511, head = cc >> 6, hd = cc & 63;
      int tok0 = m0 + s * 16 + 8 * hl;          // rows tok0..tok0+7 (r axis)
      int b = tok0 >> 12, ti0 = tok0 & (NSEQ - 1);
      size_t bh = (size_t)b * NHEAD + head;
      if (seg == 2) {                            // V^T: token axis contiguous
        v8h o;
#pragma unroll
        for (int r = 0; r < 8; r++) o[r] = (_Float16)(acc[s][nt][r] + bias[c]);
        *(v8h*)(VTb + (bh * HD + hd) * NSEQ + ti0) = o;
      } else {
        _Float16* dst = (seg == 0 ? Qb : Kb) + (bh * NSEQ + ti0) * HD + hd;
#pragma unroll
        for (int r = 0; r < 8; r++)
          dst[(size_t)r * HD] = (_Float16)(acc[s][nt][r] + bias[c]);
      }
    }
  }
}

// ---------------------------------------------------------------------------
// Flash attention (transposed): per wave one (b,h,qtile of 16 queries).
// S^T = K_tile @ Q^T, online softmax per-lane (lane column = query),
// ctx^T += V^T_tile @ P^T.  32 keys per iteration, 8 WMMAs/iter.
// ---------------------------------------------------------------------------
__global__ void flash_attn_kernel(const _Float16* __restrict__ Q,
                                  const _Float16* __restrict__ K,
                                  const _Float16* __restrict__ VT,
                                  _Float16* __restrict__ CTX) {
  int wave = blockIdx.x * 8 + (threadIdx.x >> 5);
  int qt = wave & 255;           // N/16 = 256 q-tiles
  int bh = wave >> 8;            // [0,16)
  int lane = threadIdx.x & 31, hl = lane >> 4, ln = lane & 15;
  const float scale = 0.04419417382415922f;   // 1/sqrt(512), faithful to source

  const _Float16* Qp = Q  + ((size_t)bh * NSEQ + qt * 16) * HD;
  const _Float16* Kp = K  + (size_t)bh * NSEQ * HD;
  const _Float16* Vp = VT + (size_t)bh * HD * NSEQ;

  // B-fragments of Q^T (hd x 16 queries), loaded once: col = query ln.
  v16h bqa = *(const v16h*)(Qp + (size_t)ln * HD + hl * 16);        // hd 0..31
  v16h bqb = *(const v16h*)(Qp + (size_t)ln * HD + 32 + hl * 16);   // hd 32..63

  v8f acc[4] = {};               // ctx^T: rows hd = t*16 + r + 8*hl, col query ln
  float m_i = -1e30f, l_i = 0.0f;

  for (int kk = 0; kk < NSEQ; kk += 32) {
    // score tiles S^T (keys x queries), A = K rows (key tokens)
    const _Float16* k0 = Kp + (size_t)(kk + ln) * HD;
    const _Float16* k1 = Kp + (size_t)(kk + 16 + ln) * HD;
    v8f st0 = {}, st1 = {};
    st0 = wmma_f16(load_a_frag(k0, 0, hl),  bqa, st0);
    st0 = wmma_f16(load_a_frag(k0, 32, hl), bqb, st0);
    st1 = wmma_f16(load_a_frag(k1, 0, hl),  bqa, st1);
    st1 = wmma_f16(load_a_frag(k1, 32, hl), bqb, st1);

    // online softmax over key axis (rows + lane-half), per-lane query state
    float s0[8], s1[8], cmax = -1e30f;
#pragma unroll
    for (int r = 0; r < 8; r++) {
      s0[r] = st0[r] * scale;
      s1[r] = st1[r] * scale;
      cmax = fmaxf(cmax, fmaxf(s0[r], s1[r]));
    }
    cmax = fmaxf(cmax, __shfl_xor(cmax, 16));
    float nm    = fmaxf(m_i, cmax);
    float alpha = __expf(m_i - nm);
    float p0[8], p1[8], lsum = 0.f;
#pragma unroll
    for (int r = 0; r < 8; r++) {
      p0[r] = __expf(s0[r] - nm);
      p1[r] = __expf(s1[r] - nm);
      lsum += p0[r] + p1[r];
    }
    lsum += __shfl_xor(lsum, 16);
    l_i = l_i * alpha + lsum;
    m_i = nm;
#pragma unroll
    for (int t = 0; t < 4; t++)
#pragma unroll
      for (int r = 0; r < 8; r++) acc[t][r] *= alpha;

    // Build B-fragment of P^T (32 keys x 16 queries): col = query ln,
    // K(j) = hl*16 + j; half-swap via shfl_xor(16).
    v16h pb;
#pragma unroll
    for (int r = 0; r < 8; r++) {
      float o0 = __shfl_xor(p0[r], 16);
      float o1 = __shfl_xor(p1[r], 16);
      pb[r]     = (_Float16)(hl == 0 ? p0[r] : o1);
      pb[8 + r] = (_Float16)(hl == 0 ? o0 : p1[r]);
    }

    // ctx^T += V^T @ P^T; A tiles from VT rows (contiguous over keys)
#pragma unroll
    for (int t = 0; t < 4; t++) {
      const _Float16* vrow = Vp + (size_t)(t * 16 + ln) * NSEQ + kk;
      acc[t] = wmma_f16(load_a_frag(vrow, 0, hl), pb, acc[t]);
    }
  }

  float inv = 1.0f / l_i;
  int b = bh >> 3, h = bh & 7;
  _Float16* op = CTX + ((size_t)b * NSEQ + qt * 16 + ln) * DIM + h * HD;
#pragma unroll
  for (int t = 0; t < 4; t++) {
    v8h o;
#pragma unroll
    for (int r = 0; r < 8; r++) o[r] = (_Float16)(acc[t][r] * inv);
    *(v8h*)(op + t * 16 + hl * 8) = o;   // hd = t*16 + 8*hl + r, contiguous
  }
}

// ---------------------------------------------------------------------------
// proj GEMM (32x64): ctx @ WprojT + b + x -> out (f32)
// ---------------------------------------------------------------------------
__global__ void gemm_proj_kernel(const _Float16* __restrict__ A,
                                 const _Float16* __restrict__ Wt,
                                 const float* __restrict__ bias,
                                 const float* __restrict__ X,
                                 float* __restrict__ OUT) {
  int wave = blockIdx.x * 8 + (threadIdx.x >> 5);
  const int NT = DIM / 64;   // 8
  int mt = wave / NT, n0 = (wave % NT) * 64;
  int lane = threadIdx.x & 31, hl = lane >> 4, ln = lane & 15;
  int m0 = mt * 32;
  v8f acc[2][4] = {};
  gemm_wave_32x64<DIM>(A + (size_t)(m0 + ln) * DIM,
                       A + (size_t)(m0 + 16 + ln) * DIM,
                       Wt, n0, hl, ln, acc[0], acc[1]);
#pragma unroll
  for (int s = 0; s < 2; s++) {
#pragma unroll
    for (int nt = 0; nt < 4; nt++) {
      int c = n0 + nt * 16 + ln;
#pragma unroll
      for (int r = 0; r < 8; r++) {
        int tok = m0 + s * 16 + r + 8 * hl;
        OUT[(size_t)tok * DIM + c] =
            acc[s][nt][r] + bias[c] + X[(size_t)tok * DIM + c];
      }
    }
  }
}

// ---------------------------------------------------------------------------
// FC1 (32x64): ln(out) @ W1T + b1, ReLU -> h (f16)
// ---------------------------------------------------------------------------
__global__ void gemm_fc1_kernel(const _Float16* __restrict__ A,
                                const _Float16* __restrict__ Wt,
                                const float* __restrict__ bias,
                                _Float16* __restrict__ H) {
  int wave = blockIdx.x * 8 + (threadIdx.x >> 5);
  const int NT = HID / 64;   // 32
  int mt = wave / NT, n0 = (wave % NT) * 64;
  int lane = threadIdx.x & 31, hl = lane >> 4, ln = lane & 15;
  int m0 = mt * 32;
  v8f acc[2][4] = {};
  gemm_wave_32x64<DIM>(A + (size_t)(m0 + ln) * DIM,
                       A + (size_t)(m0 + 16 + ln) * DIM,
                       Wt, n0, hl, ln, acc[0], acc[1]);
#pragma unroll
  for (int s = 0; s < 2; s++) {
#pragma unroll
    for (int nt = 0; nt < 4; nt++) {
      int c = n0 + nt * 16 + ln;
#pragma unroll
      for (int r = 0; r < 8; r++) {
        int tok = m0 + s * 16 + r + 8 * hl;
        float v = acc[s][nt][r] + bias[c];
        H[(size_t)tok * HID + c] = (_Float16)(v > 0.f ? v : 0.f);
      }
    }
  }
}

// ---------------------------------------------------------------------------
// FC2 (32x64): h @ W2T + b2, ReLU, + out -> d_out (f32)
// ---------------------------------------------------------------------------
__global__ void gemm_fc2_kernel(const _Float16* __restrict__ A,
                                const _Float16* __restrict__ Wt,
                                const float* __restrict__ bias,
                                const float* __restrict__ OUT,
                                float* __restrict__ Y) {
  int wave = blockIdx.x * 8 + (threadIdx.x >> 5);
  const int NT = DIM / 64;   // 8
  int mt = wave / NT, n0 = (wave % NT) * 64;
  int lane = threadIdx.x & 31, hl = lane >> 4, ln = lane & 15;
  int m0 = mt * 32;
  v8f acc[2][4] = {};
  gemm_wave_32x64<HID>(A + (size_t)(m0 + ln) * HID,
                       A + (size_t)(m0 + 16 + ln) * HID,
                       Wt, n0, hl, ln, acc[0], acc[1]);
#pragma unroll
  for (int s = 0; s < 2; s++) {
#pragma unroll
    for (int nt = 0; nt < 4; nt++) {
      int c = n0 + nt * 16 + ln;
#pragma unroll
      for (int r = 0; r < 8; r++) {
        int tok = m0 + s * 16 + r + 8 * hl;
        float v = acc[s][nt][r] + bias[c];
        v = v > 0.f ? v : 0.f;
        Y[(size_t)tok * DIM + c] = OUT[(size_t)tok * DIM + c] + v;
      }
    }
  }
}

// ---------------------------------------------------------------------------
// Launch: workspace layout (bytes), buffers reused once dead:
//   0         WqkvT  (1.5MB) | 1572864 WprojT | 2097152 W1T | 4194304 W2T
//   6291456   xln / out_ln (8MB f16)
//   14680064  Q (8MB)  -- overlaid by H (32MB spans Q,K,VT,CTX)
//   23068672  K (8MB)
//   31457280  VT (8MB)
//   39845888  ctx (8MB)
//   48234496  out (16MB f32)          total = 65,011,712 B
// ---------------------------------------------------------------------------
extern "C" void kernel_launch(void* const* d_in, const int* in_sizes, int n_in,
                              void* d_out, int out_size, void* d_ws, size_t ws_size,
                              hipStream_t stream) {
  (void)in_sizes; (void)n_in; (void)out_size; (void)ws_size;
  const float* x      = (const float*)d_in[0];
  const float* ln_g   = (const float*)d_in[1];
  const float* ln_b   = (const float*)d_in[2];
  const float* w_qkv  = (const float*)d_in[3];
  const float* b_qkv  = (const float*)d_in[4];
  const float* w_proj = (const float*)d_in[5];
  const float* b_proj = (const float*)d_in[6];
  const float* w1     = (const float*)d_in[7];
  const float* b1     = (const float*)d_in[8];
  const float* w2     = (const float*)d_in[9];
  const float* b2     = (const float*)d_in[10];
  float* out_final    = (float*)d_out;

  char* W = (char*)d_ws;
  _Float16* wqkvt  = (_Float16*)(W + 0);
  _Float16* wprojt = (_Float16*)(W + 1572864);
  _Float16* w1t    = (_Float16*)(W + 2097152);
  _Float16* w2t    = (_Float16*)(W + 4194304);
  _Float16* xln    = (_Float16*)(W + 6291456);    // reused as out_ln
  _Float16* Qb     = (_Float16*)(W + 14680064);
  _Float16* Kb     = (_Float16*)(W + 23068672);
  _Float16* VTb    = (_Float16*)(W + 31457280);
  _Float16* ctx    = (_Float16*)(W + 39845888);
  float*    outb   = (float*)   (W + 48234496);
  _Float16* hbuf   = (_Float16*)(W + 14680064);   // overlays Q..ctx (32MB)

  convert_w_kernel<<<12288, 256, 0, stream>>>(w_qkv, w_proj, w1, w2,
                                              wqkvt, wprojt, w1t, w2t);
  ln_kernel<<<ROWS / 8, 256, 0, stream>>>(x, ln_g, ln_b, xln);
  gemm_qkv_kernel<<<(ROWS / 32) * (3 * DIM / 64) / 8, 256, 0, stream>>>(
      xln, wqkvt, b_qkv, Qb, Kb, VTb);
  flash_attn_kernel<<<(NB * NHEAD * (NSEQ / 16)) / 8, 256, 0, stream>>>(
      Qb, Kb, VTb, ctx);
  gemm_proj_kernel<<<(ROWS / 32) * (DIM / 64) / 8, 256, 0, stream>>>(
      ctx, wprojt, b_proj, x, outb);
  ln_kernel<<<ROWS / 8, 256, 0, stream>>>(outb, ln_g, ln_b, xln);
  gemm_fc1_kernel<<<(ROWS / 32) * (HID / 64) / 8, 256, 0, stream>>>(
      xln, w1t, b1, hbuf);
  gemm_fc2_kernel<<<(ROWS / 32) * (DIM / 64) / 8, 256, 0, stream>>>(
      hbuf, w2t, b2, outb, out_final);
}
